// ContrastiveProjectionHead_8821862826752
// MI455X (gfx1250) — matrix operand order repaired and checked by reading, not compiled, not run on hardware
//
#include <hip/hip_runtime.h>

// ---------------------------------------------------------------------------
// ContrastiveProjectionHead for MI455X (gfx1250, wave32, WMMA f16 16x16x32)
//
//   z'[16384,512] = relu(x'[16384,1024]) @ w[1024,512] + b      (phase 1, WMMA)
//   zn[8192,1024] = normalize(z reshaped)                        (phase 2)
//   nll[i] = -pos_i + logsumexp_j( mask(zn@zn^T)[i,j] / 0.1 )    (phase 3, WMMA)
//   out    = mean(nll)                                           (phase 4)
//
// Phase 3 (137 GFLOP) dominates; zn(f16)=16MB is L2-resident, so it is pure
// matrix-core work. logsumexp uses a fixed shift of 10 (cos/T <= 10).
// ---------------------------------------------------------------------------

typedef __attribute__((ext_vector_type(16))) _Float16 v16h;
typedef __attribute__((ext_vector_type(8)))  float    v8f;
typedef __attribute__((ext_vector_type(4)))  _Float16 v4h;
typedef __attribute__((ext_vector_type(4)))  int      v4i;

union Frag32 { v16h h; uint4 q[2]; };

// A-matrix 16x32 f16 fragment (documented layout):
//   lane L: row = base + (L&15); halves [0..7] at k0+8*(L>>4), [8..15] at k0+16+8*(L>>4)
__device__ __forceinline__ v16h load_a_f16(const _Float16* __restrict__ p, int hh) {
  Frag32 f;
  f.q[0] = *(const uint4*)(p + 8 * hh);
  f.q[1] = *(const uint4*)(p + 16 + 8 * hh);
  return f.h;
}

// B-matrix 32x16 f16 fragment: lane L: column = L&15, halves [0..7] at k0+8*(L>>4),
// [8..15] at k0+16+8*(L>>4)   (B rows striped across lanes)
__device__ __forceinline__ v16h load_b_f16(const _Float16* __restrict__ p, int hh) {
  Frag32 f;
  f.q[0] = *(const uint4*)(p + 8 * hh);
  f.q[1] = *(const uint4*)(p + 16 + 8 * hh);
  return f.h;
}

// A fragment from fp32 source with fused relu + f16 convert (phase 1).
__device__ __forceinline__ v16h load_a_relu_f32(const float* __restrict__ p, int hh) {
  const float4* s0 = (const float4*)(p + 8 * hh);
  const float4* s1 = (const float4*)(p + 16 + 8 * hh);
  float4 a = s0[0], b = s0[1], c = s1[0], d = s1[1];
  v16h r;
  r[0]  = (_Float16)fmaxf(a.x, 0.f); r[1]  = (_Float16)fmaxf(a.y, 0.f);
  r[2]  = (_Float16)fmaxf(a.z, 0.f); r[3]  = (_Float16)fmaxf(a.w, 0.f);
  r[4]  = (_Float16)fmaxf(b.x, 0.f); r[5]  = (_Float16)fmaxf(b.y, 0.f);
  r[6]  = (_Float16)fmaxf(b.z, 0.f); r[7]  = (_Float16)fmaxf(b.w, 0.f);
  r[8]  = (_Float16)fmaxf(c.x, 0.f); r[9]  = (_Float16)fmaxf(c.y, 0.f);
  r[10] = (_Float16)fmaxf(c.z, 0.f); r[11] = (_Float16)fmaxf(c.w, 0.f);
  r[12] = (_Float16)fmaxf(d.x, 0.f); r[13] = (_Float16)fmaxf(d.y, 0.f);
  r[14] = (_Float16)fmaxf(d.z, 0.f); r[15] = (_Float16)fmaxf(d.w, 0.f);
  return r;
}

#define WMMA_F16(a, b, c) \
  __builtin_amdgcn_wmma_f32_16x16x32_f16(false, (a), false, (b), (short)0, (c), false, false)

// ---------------------------------------------------------------------------
// Async global -> LDS staging (CDNA5 GLOBAL_LOAD_ASYNC_TO_LDS_B128, ASYNCcnt).
// Builtin signature (from hipcc diagnostic): (int4 global*, int4 shared*, imm, imm).
// Pointers built via integer round-trip: flat global addr == global addr; flat LDS
// addr low 32 bits == wave-relative LDS byte offset (ISA 10.2).
#if defined(__gfx1250__) && __has_builtin(__builtin_amdgcn_global_load_async_to_lds_b128)
#define HAVE_ASYNC_LDS 1
#else
#define HAVE_ASYNC_LDS 0
#endif

typedef __attribute__((address_space(1))) v4i* as1_v4i;
typedef __attribute__((address_space(3))) v4i* as3_v4i;

__device__ __forceinline__ void copy16_to_lds(_Float16* l, const _Float16* g) {
#if HAVE_ASYNC_LDS
  __builtin_amdgcn_global_load_async_to_lds_b128(
      (as1_v4i)(uintptr_t)g, (as3_v4i)(uint32_t)(uintptr_t)l, 0, 0);
#else
  *(uint4*)l = *(const uint4*)g;
#endif
}

__device__ __forceinline__ void wait_stage() {
#if HAVE_ASYNC_LDS
#if __has_builtin(__builtin_amdgcn_s_wait_asynccnt)
  __builtin_amdgcn_s_wait_asynccnt(0);
#else
  asm volatile("s_wait_asynccnt 0x0" ::: "memory");
#endif
#endif
}

// ---------------------------------------------------------------------------
// Kernel 0: transpose w[1024,512] f32 -> wT[512,1024] f16 (so B frags are row loads)
__global__ void wtrans_kernel(const float* __restrict__ w, _Float16* __restrict__ wT) {
  int idx = blockIdx.x * 256 + threadIdx.x;       // 512*1024 elements
  int n = idx >> 10, k = idx & 1023;
  wT[idx] = (_Float16)w[k * 512 + n];
}

// ---------------------------------------------------------------------------
// Kernel 1: z'[16384,512] = relu(x'[16384,1024]) @ w + b ; 128x128 block tile,
// 8 waves, each wave = 16 rows x 128 cols = 8 WMMA accumulators, K loop of 32.
// All 8 B fragments are preloaded per k-step so loads batch into one clause
// and the 8 WMMAs issue back-to-back.
__global__ __launch_bounds__(256) void proj_kernel(const float* __restrict__ x,
                                                   const _Float16* __restrict__ wT,
                                                   const float* __restrict__ bias,
                                                   float* __restrict__ z) {
  const int nb = blockIdx.x & 3, mb = blockIdx.x >> 2;
  const int m0 = mb * 128, n0 = nb * 128;
  const int tid = threadIdx.x, wv = tid >> 5, lane = tid & 31;
  const int hh = lane >> 4, ln = lane & 15;

  v8f acc[8] = {};
  const float* xrow = x + (size_t)(m0 + 16 * wv + ln) * 1024;
  const _Float16* wrow = wT + (size_t)(n0 + ln) * 1024;   // fragment f adds 16*f rows

  for (int k0 = 0; k0 < 1024; k0 += 32) {
    v16h bf[8];
#pragma unroll
    for (int f = 0; f < 8; ++f)
      bf[f] = load_b_f16(wrow + (size_t)(16 * f) * 1024 + k0, hh);
    v16h a = load_a_relu_f32(xrow + k0, hh);
#pragma unroll
    for (int f = 0; f < 8; ++f)
      acc[f] = WMMA_F16(a, bf[f], acc[f]);
  }

  // C layout: VGPR v, lane L -> row m0+16*wv + v + 8*(L>>4), col n0+16*f + (L&15)
  const int rowC = m0 + 16 * wv + 8 * hh;
#pragma unroll
  for (int f = 0; f < 8; ++f) {
    const int col = n0 + 16 * f + ln;
    const float bv = bias[col];
#pragma unroll
    for (int v = 0; v < 8; ++v)
      z[(size_t)(rowC + v) * 512 + col] = acc[f][v] + bv;
  }
}

// ---------------------------------------------------------------------------
// Kernel 2: per-row L2 normalize z[8192,1024] f32 -> zn f16
__global__ __launch_bounds__(256) void normalize_kernel(const float* __restrict__ z,
                                                        _Float16* __restrict__ zn) {
  const int row = blockIdx.x;
  const float4 v = ((const float4*)(z + (size_t)row * 1024))[threadIdx.x];
  float s = v.x * v.x + v.y * v.y + v.z * v.z + v.w * v.w;
#pragma unroll
  for (int m = 1; m < 32; m <<= 1) s += __shfl_xor(s, m);
  __shared__ float wsum[8];
  if ((threadIdx.x & 31) == 0) wsum[threadIdx.x >> 5] = s;
  __syncthreads();
  float tot = 0.f;
#pragma unroll
  for (int i = 0; i < 8; ++i) tot += wsum[i];
  const float inv = 1.f / fmaxf(sqrtf(tot), 1e-8f);
  v4h h;
  h[0] = (_Float16)(v.x * inv); h[1] = (_Float16)(v.y * inv);
  h[2] = (_Float16)(v.z * inv); h[3] = (_Float16)(v.w * inv);
  *(v4h*)(zn + (size_t)row * 1024 + threadIdx.x * 4) = h;
}

// ---------------------------------------------------------------------------
// Kernel 3: similarity + streaming sum-of-exp.
// Block = 128 rows x 512 cols (4 N-tiles of 128), 8 waves; wave = 16 rows.
// B strip (128 rows x 32 K) double-buffered in LDS via async copy, padded
// stride 40 halves (80B) for conflict-free ds_load_b128.
#define BSTRIDE 40
__device__ __forceinline__ void stageB(_Float16* __restrict__ dst,
                                       const _Float16* __restrict__ zn,
                                       int j0, int k0, int tid) {
#pragma unroll
  for (int it = 0; it < 2; ++it) {
    const int idx = tid + it * 256;    // 512 x uint4 = 128 rows x 32 halves
    const int row = idx >> 2, seg = idx & 3;
    copy16_to_lds(dst + row * BSTRIDE + seg * 8,
                  zn + (size_t)(j0 + row) * 1024 + k0 + seg * 8);
  }
}

__global__ __launch_bounds__(256) void sim_lse_kernel(const _Float16* __restrict__ zn,
                                                      float* __restrict__ psum,
                                                      float* __restrict__ posArr) {
  __shared__ _Float16 Bs[2][128 * BSTRIDE];   // 20 KB
  const int m0 = blockIdx.x * 128;
  const int jsplit = blockIdx.y;              // 16 column splits of 512
  const int tid = threadIdx.x, wv = tid >> 5, lane = tid & 31;
  const int hh = lane >> 4, ln = lane & 15;

  const int posTile = (m0 + 4096) & 8191;     // (i - B/2) mod B == (i + B/2) mod B
  const bool writesPos = (posTile >> 9) == jsplit;

  float rowAcc[8] = {0.f, 0.f, 0.f, 0.f, 0.f, 0.f, 0.f, 0.f};
  float posv = 0.f;
  const _Float16* arow = zn + (size_t)(m0 + 16 * wv + ln) * 1024;

  for (int t = 0; t < 4; ++t) {
    const int j0 = jsplit * 512 + t * 128;
    v8f acc[8] = {};

    stageB(Bs[0], zn, j0, 0, tid);
    wait_stage();
    __syncthreads();
    for (int kc = 0; kc < 32; ++kc) {
      if (kc + 1 < 32) stageB(Bs[(kc + 1) & 1], zn, j0, (kc + 1) * 32, tid);
      // preload all 8 B fragments, then issue 8 WMMAs back-to-back
      const _Float16* bbase = Bs[kc & 1];
      v16h bf[8];
#pragma unroll
      for (int f = 0; f < 8; ++f)
        bf[f] = load_b_f16(bbase + (16 * f + ln) * BSTRIDE, hh);
      v16h a = load_a_f16(arow + kc * 32, hh);
#pragma unroll
      for (int f = 0; f < 8; ++f)
        acc[f] = WMMA_F16(a, bf[f], acc[f]);
      wait_stage();       // waits only for chunk kc+1 (overlapped with WMMAs above)
      __syncthreads();
    }

    // epilogue: scale by 1/T=10, mask diagonal, capture pos, accumulate exp(s-10)
    const bool diagT = (j0 == m0);
    const bool posT  = (j0 == posTile);
#pragma unroll
    for (int f = 0; f < 8; ++f) {
#pragma unroll
      for (int v = 0; v < 8; ++v) {
        float s = acc[f][v] * 10.f;
        // fragment-diagonal element (row == col within tile):
        const bool isD = (f == wv) &&
                         ((lane < 8 && v == lane) || (lane >= 24 && v == lane - 24));
        if (posT && isD) posv = s;
        if (diagT && isD) s = -1e30f;   // NEG_INF/T -> exp == 0
        rowAcc[v] += __expf(s - 10.f);
      }
    }
  }

  // reduce each row's partial across its 16-lane group (xor 1,2,4,8 stays in-group)
  float mySum = 0.f;
#pragma unroll
  for (int v = 0; v < 8; ++v) {
    float s = rowAcc[v];
    s += __shfl_xor(s, 1); s += __shfl_xor(s, 2);
    s += __shfl_xor(s, 4); s += __shfl_xor(s, 8);
    if ((lane & 7) == v) mySum = s;
  }
  // writer lanes: 0..7 -> rows +0..7 (hh=0), 24..31 -> rows +8..15 (hh=1)
  if (lane < 8) {
    const int row = m0 + 16 * wv + lane;
    psum[jsplit * 8192 + row] = mySum;
    if (writesPos) posArr[row] = posv;
  } else if (lane >= 24) {
    const int row = m0 + 16 * wv + 8 + (lane - 24);
    psum[jsplit * 8192 + row] = mySum;
    if (writesPos) posArr[row] = posv;
  }
}

// ---------------------------------------------------------------------------
// Kernel 4: combine 16 column-split partials -> nll per row
__global__ void combine_kernel(const float* __restrict__ psum,
                               const float* __restrict__ posArr,
                               float* __restrict__ nll) {
  const int row = blockIdx.x * 256 + threadIdx.x;   // 8192 rows
  float s = 0.f;
#pragma unroll
  for (int p = 0; p < 16; ++p) s += psum[p * 8192 + row];
  nll[row] = -posArr[row] + 10.f + logf(s);
}

// Kernel 5: deterministic mean over 8192 rows
__global__ void mean_kernel(const float* __restrict__ nll, float* __restrict__ out) {
  __shared__ float sb[256];
  float s = 0.f;
  for (int i = threadIdx.x; i < 8192; i += 256) s += nll[i];
  sb[threadIdx.x] = s;
  __syncthreads();
  for (int off = 128; off > 0; off >>= 1) {
    if (threadIdx.x < off) sb[threadIdx.x] += sb[threadIdx.x + off];
    __syncthreads();
  }
  if (threadIdx.x == 0) out[0] = sb[0] * (1.f / 8192.f);
}

// ---------------------------------------------------------------------------
extern "C" void kernel_launch(void* const* d_in, const int* in_sizes, int n_in,
                              void* d_out, int out_size, void* d_ws, size_t ws_size,
                              hipStream_t stream) {
  const float* x = (const float*)d_in[0];   // [8192, 2048]
  const float* w = (const float*)d_in[1];   // [1024, 512]
  const float* b = (const float*)d_in[2];   // [512]
  float* out = (float*)d_out;

  char* p = (char*)d_ws;
  float*    z    = (float*)p;                 p += (size_t)16384 * 512 * 4;  // 32 MB
  _Float16* zn   = (_Float16*)p;              p += (size_t)8192 * 1024 * 2;  // 16 MB
  _Float16* wT   = (_Float16*)p;              p += (size_t)512 * 1024 * 2;   //  1 MB
  float*    psum = (float*)p;                 p += (size_t)16 * 8192 * 4;    // 512 KB
  float*    posA = (float*)p;                 p += (size_t)8192 * 4;
  float*    nll  = (float*)p;                 p += (size_t)8192 * 4;

  wtrans_kernel<<<2048, 256, 0, stream>>>(w, wT);
  proj_kernel<<<512, 256, 0, stream>>>(x, wT, b, z);          // 128 M-blk x 4 N-blk
  normalize_kernel<<<8192, 256, 0, stream>>>(z, zn);
  sim_lse_kernel<<<dim3(64, 16), 256, 0, stream>>>(zn, psum, posA);
  combine_kernel<<<32, 256, 0, stream>>>(psum, posA, nll);
  mean_kernel<<<1, 256, 0, stream>>>(nll, out);
}
